// Model_4157528343216
// MI455X (gfx1250) — matrix operand order repaired and checked by reading, not compiled, not run on hardware
//
#include <hip/hip_runtime.h>

// ---------------------------------------------------------------------------
// CDNA5 (gfx1250, wave32) implementation of 2-layer bipartite GraphSAGE +
// edge classifier. Dense transforms use v_wmma_f32_16x16x32_bf16 (f32 accum)
// with weight B-panels staged in LDS; sparse aggregation uses hardware f32
// global atomics (memory-roofline-bound part: ~4-5 GB @ 23.3 TB/s).
// ---------------------------------------------------------------------------

typedef __attribute__((ext_vector_type(16))) __bf16 v16bf;
typedef __attribute__((ext_vector_type(8)))  float  v8f;

#define D_FEAT 128

static inline int cdiv_i(long long a, long long b) { return (int)((a + b - 1) / b); }

// ---------------------------------------------------------------------------
// small helpers / kernels
// ---------------------------------------------------------------------------
__global__ void fill0_kernel(float* __restrict__ p, size_t n) {
  size_t i  = (size_t)blockIdx.x * blockDim.x + threadIdx.x;
  size_t st = (size_t)gridDim.x * blockDim.x;
  for (; i < n; i += st) p[i] = 0.0f;
}

// convert + transpose weight:  W[K][N] f32  ->  WT[N][K] bf16
__global__ void prep_w_kernel(const float* __restrict__ W, __bf16* __restrict__ WT,
                              int K, int N) {
  int i = blockIdx.x * blockDim.x + threadIdx.x;
  if (i < K * N) {
    int k = i / N, n = i % N;
    WT[(size_t)n * K + k] = (__bf16)W[i];
  }
}

// degree counts: cnt_dis over edge_dst, cnt_drug over edge_src
__global__ void count_deg_kernel(const int* __restrict__ esrc, const int* __restrict__ edst,
                                 float* __restrict__ cnt_drug, float* __restrict__ cnt_dis,
                                 int E) {
  int e = blockIdx.x * blockDim.x + threadIdx.x;
  if (e < E) {
    unsafeAtomicAdd(cnt_drug + esrc[e], 1.0f);
    unsafeAtomicAdd(cnt_dis  + edst[e], 1.0f);
  }
}

// scatter-add: agg[scatter_idx[e]][:] += x[gather_idx[e]][:]
// 32 threads per edge, 4 channels (float4) per thread: 512B coalesced per edge.
__global__ void scatter_add_kernel(const float* __restrict__ x,
                                   const int* __restrict__ gidx,
                                   const int* __restrict__ sidx,
                                   float* __restrict__ agg, long long E) {
  long long t = (long long)blockIdx.x * blockDim.x + threadIdx.x;
  long long e = t >> 5;
  if (e >= E) return;
  int c = (int)(t & 31) * 4;
  int g = gidx[e], s = sidx[e];
  float4 v = *reinterpret_cast<const float4*>(x + (size_t)g * D_FEAT + c);
  float* dp = agg + (size_t)s * D_FEAT + c;
  unsafeAtomicAdd(dp + 0, v.x);
  unsafeAtomicAdd(dp + 1, v.y);
  unsafeAtomicAdd(dp + 2, v.z);
  unsafeAtomicAdd(dp + 3, v.w);
}

// ---------------------------------------------------------------------------
// A-operand pack: per the 16-bit A layout (16x32), lane L holds row M=L%16 and
// K = {base..base+7} u {16+base..16+base+7}, base = (L<16 ? 0 : 8): two
// contiguous 8-float runs, scaled then converted to bf16.
// ---------------------------------------------------------------------------
__device__ __forceinline__ v16bf pack_a(const float* __restrict__ p, int base, float s) {
  const float4* q = reinterpret_cast<const float4*>(p);
  int i0 = base >> 2;                     // 0 or 2
  float4 f0 = q[i0],     f1 = q[i0 + 1];  // K = base..base+7
  float4 f2 = q[i0 + 4], f3 = q[i0 + 5];  // K = 16+base..16+base+7
  v16bf a;
  a[0]=(__bf16)(f0.x*s); a[1]=(__bf16)(f0.y*s); a[2]=(__bf16)(f0.z*s); a[3]=(__bf16)(f0.w*s);
  a[4]=(__bf16)(f1.x*s); a[5]=(__bf16)(f1.y*s); a[6]=(__bf16)(f1.z*s); a[7]=(__bf16)(f1.w*s);
  a[8]=(__bf16)(f2.x*s); a[9]=(__bf16)(f2.y*s); a[10]=(__bf16)(f2.z*s); a[11]=(__bf16)(f2.w*s);
  a[12]=(__bf16)(f3.x*s); a[13]=(__bf16)(f3.y*s); a[14]=(__bf16)(f3.z*s); a[15]=(__bf16)(f3.w*s);
  return a;
}

// ---------------------------------------------------------------------------
// SAGE transform: out = act( (agg/max(cnt,1)) @ Wl + b + xdst @ Wr )
// Weight panels (2 x 32KB bf16) staged to LDS once per block; each wave then
// computes a 16-row x 128-col tile with 8 f32 WMMA accumulators, K = 4 x 32.
// ---------------------------------------------------------------------------
__global__ __launch_bounds__(128) void sage_gemm_kernel(
    const float* __restrict__ agg, const float* __restrict__ cnt,
    const float* __restrict__ xdst,
    const __bf16* __restrict__ WlT, const __bf16* __restrict__ WrT,
    const float* __restrict__ bias, float* __restrict__ out,
    int n, int do_relu) {
  __shared__ __align__(32) __bf16 sWl[D_FEAT * D_FEAT];  // 32KB
  __shared__ __align__(32) __bf16 sWr[D_FEAT * D_FEAT];  // 32KB

  // ---- stage both B panels into LDS (b128 global loads -> b128 ds stores) ----
  {
    const uint4* gl = reinterpret_cast<const uint4*>(WlT);
    const uint4* gr = reinterpret_cast<const uint4*>(WrT);
    uint4* sl = reinterpret_cast<uint4*>(sWl);
    uint4* sr = reinterpret_cast<uint4*>(sWr);
    const int NV = D_FEAT * D_FEAT / 8;  // 2048 uint4 per panel
#pragma unroll
    for (int i = threadIdx.x; i < NV; i += 128) {
      sl[i] = gl[i];
      sr[i] = gr[i];
    }
  }
  __syncthreads();

  int wave = threadIdx.x >> 5;
  int lane = threadIdx.x & 31;
  int m    = lane & 15;           // output col within 16-tile / A row within tile
  int hi   = lane >> 4;           // half-wave selector
  int rowbase = (blockIdx.x * 4 + wave) * 16;
  int row  = rowbase + m;
  int rowc = row < n ? row : n - 1;              // clamp (EXEC must stay full for WMMA)
  float s = 1.0f / fmaxf(cnt[rowc], 1.0f);
  const float* aggp = agg  + (size_t)rowc * D_FEAT;
  const float* xp   = xdst + (size_t)rowc * D_FEAT;
  int abase = hi * 8, hioff = hi * 8;

  v8f acc[8] = {};
#pragma unroll
  for (int k0 = 0; k0 < D_FEAT; k0 += 32) {
    v16bf Am = pack_a(aggp + k0, abase, s);      // mean aggregate (scaled on load)
    v16bf As = pack_a(xp   + k0, abase, 1.0f);   // self features
#pragma unroll
    for (int t = 0; t < 8; ++t) {
      const __bf16* bl = sWl + (t * 16 + m) * D_FEAT + k0 + hioff;
      const __bf16* br = sWr + (t * 16 + m) * D_FEAT + k0 + hioff;
      v16bf Bl, Br;
#pragma unroll
      for (int q = 0; q < 16; ++q) { Bl[q] = bl[q < 8 ? q : q + 8]; Br[q] = br[q < 8 ? q : q + 8]; }
      acc[t] = __builtin_amdgcn_wmma_f32_16x16x32_bf16(false, Am, false, Bl,
                                                       (short)0, acc[t], false, false);
      acc[t] = __builtin_amdgcn_wmma_f32_16x16x32_bf16(false, As, false, Br,
                                                       (short)0, acc[t], false, false);
    }
  }
  // epilogue: bias (+ReLU), store (coalesced over N across lanes 0..15)
#pragma unroll
  for (int t = 0; t < 8; ++t) {
    float bv = bias[t * 16 + m];
#pragma unroll
    for (int r = 0; r < 8; ++r) {
      int orow = rowbase + r + hi * 8;
      float v = acc[t][r] + bv;
      if (do_relu) v = fmaxf(v, 0.0f);
      if (orow < n) out[(size_t)orow * D_FEAT + t * 16 + m] = v;
    }
  }
}

// ---------------------------------------------------------------------------
// Fused edge classifier:
//   e = [z_drug2[ls] || z_dis2[ld]]  (K=256)
//   h = relu(e @ Wc1 + bc1)          (WMMA, 16 edges x 128 per wave; Wc1T in LDS)
//   out = h @ Wc2 + bc2              (per-lane partials + shfl_xor reduction)
// ---------------------------------------------------------------------------
__global__ __launch_bounds__(128) void edge_cls_kernel(
    const float* __restrict__ zdrug, const float* __restrict__ zdis,
    const int* __restrict__ ls, const int* __restrict__ ld,
    const __bf16* __restrict__ Wc1T, const float* __restrict__ bc1,
    const float* __restrict__ Wc2, const float* __restrict__ bc2,
    float* __restrict__ out, int EL) {
  __shared__ __align__(32) __bf16 sW[D_FEAT * 256];  // 64KB

  {
    const uint4* g = reinterpret_cast<const uint4*>(Wc1T);
    uint4* sp = reinterpret_cast<uint4*>(sW);
    const int NV = D_FEAT * 256 / 8;  // 4096 uint4
#pragma unroll
    for (int i = threadIdx.x; i < NV; i += 128) sp[i] = g[i];
  }
  __syncthreads();

  int wave = threadIdx.x >> 5;
  int lane = threadIdx.x & 31;
  int m = lane & 15, hi = lane >> 4;
  int ebase = (blockIdx.x * 4 + wave) * 16;
  int e  = ebase + m;
  int ec = e < EL ? e : EL - 1;
  const float* ps = zdrug + (size_t)ls[ec] * D_FEAT;
  const float* pd = zdis  + (size_t)ld[ec] * D_FEAT;
  int abase = hi * 8, hioff = hi * 8;

  v8f acc[8] = {};
#pragma unroll
  for (int k0 = 0; k0 < 256; k0 += 32) {
    const float* src = (k0 < D_FEAT) ? (ps + k0) : (pd + (k0 - D_FEAT));
    v16bf A = pack_a(src, abase, 1.0f);
#pragma unroll
    for (int t = 0; t < 8; ++t) {
      const __bf16* bp = sW + (t * 16 + m) * 256 + k0 + hioff;
      v16bf B;
#pragma unroll
      for (int q = 0; q < 16; ++q) B[q] = bp[q < 8 ? q : q + 8];
      acc[t] = __builtin_amdgcn_wmma_f32_16x16x32_bf16(false, A, false, B,
                                                       (short)0, acc[t], false, false);
    }
  }
  // second layer: p[r] = sum_N relu(h[M][N]+bc1[N]) * Wc2[N], M = r + hi*8
  float p[8] = {0, 0, 0, 0, 0, 0, 0, 0};
#pragma unroll
  for (int t = 0; t < 8; ++t) {
    float bv = bc1[t * 16 + m];
    float w2 = Wc2[t * 16 + m];
#pragma unroll
    for (int r = 0; r < 8; ++r)
      p[r] += fmaxf(acc[t][r] + bv, 0.0f) * w2;
  }
#pragma unroll
  for (int r = 0; r < 8; ++r) {
    p[r] += __shfl_xor(p[r], 8);
    p[r] += __shfl_xor(p[r], 4);
    p[r] += __shfl_xor(p[r], 2);
    p[r] += __shfl_xor(p[r], 1);
  }
  if (m == 0) {
    float b2 = bc2[0];
#pragma unroll
    for (int r = 0; r < 8; ++r) {
      int orow = ebase + r + hi * 8;
      if (orow < EL) out[orow] = p[r] + b2;
    }
  }
}

// ---------------------------------------------------------------------------
// host side
// ---------------------------------------------------------------------------
extern "C" void kernel_launch(void* const* d_in, const int* in_sizes, int n_in,
                              void* d_out, int out_size, void* d_ws, size_t ws_size,
                              hipStream_t stream) {
  const float* x_drug = (const float*)d_in[0];
  const float* x_dis  = (const float*)d_in[1];
  const int* e_src = (const int*)d_in[2];
  const int* e_dst = (const int*)d_in[3];
  const int* l_src = (const int*)d_in[4];
  const int* l_dst = (const int*)d_in[5];
  const float* W1tdl = (const float*)d_in[6];  const float* b1td = (const float*)d_in[7];
  const float* W1tdr = (const float*)d_in[8];
  const float* W1dtl = (const float*)d_in[9];  const float* b1dt = (const float*)d_in[10];
  const float* W1dtr = (const float*)d_in[11];
  const float* W2tdl = (const float*)d_in[12]; const float* b2td = (const float*)d_in[13];
  const float* W2tdr = (const float*)d_in[14];
  const float* W2dtl = (const float*)d_in[15]; const float* b2dt = (const float*)d_in[16];
  const float* W2dtr = (const float*)d_in[17];
  const float* Wc1 = (const float*)d_in[18]; const float* bc1 = (const float*)d_in[19];
  const float* Wc2 = (const float*)d_in[20]; const float* bc2 = (const float*)d_in[21];

  const int nDrug = in_sizes[0] / D_FEAT;        // 50000
  const int nDis  = in_sizes[1] / D_FEAT;        // 20000
  const long long E = in_sizes[2];               // 1,000,000
  const int EL = in_sizes[4];                    // 200,000

  // ---- workspace carve-out (256B aligned slices) ----
  char* w = (char*)d_ws;
  size_t off = 0;
  auto carve = [&](size_t bytes) -> char* {
    char* p = w + off;
    off = (off + bytes + 255) & ~(size_t)255;
    return p;
  };
  __bf16* WT[9];
  const float* Wsrc[9] = {W1tdl, W1tdr, W1dtl, W1dtr, W2tdl, W2tdr, W2dtl, W2dtr, Wc1};
  const int    Wk[9]   = {128, 128, 128, 128, 128, 128, 128, 128, 256};
  for (int i = 0; i < 9; ++i) WT[i] = (__bf16*)carve((size_t)Wk[i] * 128 * 2);

  float* cnt_dis  = (float*)carve((size_t)nDis  * 4);
  float* cnt_drug = (float*)carve((size_t)nDrug * 4);
  float* agg_dis  = (float*)carve((size_t)nDis  * D_FEAT * 4);
  float* agg_drug = (float*)carve((size_t)nDrug * D_FEAT * 4);
  float* z_dis    = (float*)carve((size_t)nDis  * D_FEAT * 4);
  float* z_drug   = (float*)carve((size_t)nDrug * D_FEAT * 4);
  float* z_dis2   = (float*)carve((size_t)nDis  * D_FEAT * 4);
  float* z_drug2  = (float*)carve((size_t)nDrug * D_FEAT * 4);
  (void)ws_size; (void)n_in; (void)out_size;

  // ---- weight conversion (bf16, transposed for contiguous B loads) ----
  for (int i = 0; i < 9; ++i) {
    int tot = Wk[i] * 128;
    prep_w_kernel<<<cdiv_i(tot, 256), 256, 0, stream>>>(Wsrc[i], WT[i], Wk[i], 128);
  }

  // ---- zero counters + layer-1 aggregation buffers ----
  fill0_kernel<<<256, 256, 0, stream>>>(cnt_dis,  (size_t)nDis);
  fill0_kernel<<<256, 256, 0, stream>>>(cnt_drug, (size_t)nDrug);
  fill0_kernel<<<2048, 256, 0, stream>>>(agg_dis,  (size_t)nDis  * D_FEAT);
  fill0_kernel<<<2048, 256, 0, stream>>>(agg_drug, (size_t)nDrug * D_FEAT);

  count_deg_kernel<<<cdiv_i(E, 256), 256, 0, stream>>>(e_src, e_dst, cnt_drug, cnt_dis, (int)E);

  // ---- layer 1 ----
  scatter_add_kernel<<<cdiv_i(E * 32, 256), 256, 0, stream>>>(x_drug, e_src, e_dst, agg_dis, E);
  sage_gemm_kernel<<<cdiv_i(nDis, 64), 128, 0, stream>>>(
      agg_dis, cnt_dis, x_dis, WT[0], WT[1], b1td, z_dis, nDis, 1);

  scatter_add_kernel<<<cdiv_i(E * 32, 256), 256, 0, stream>>>(x_dis, e_dst, e_src, agg_drug, E);
  sage_gemm_kernel<<<cdiv_i(nDrug, 64), 128, 0, stream>>>(
      agg_drug, cnt_drug, x_drug, WT[2], WT[3], b1dt, z_drug, nDrug, 1);

  // ---- layer 2 (reuse aggregation buffers) ----
  fill0_kernel<<<2048, 256, 0, stream>>>(agg_dis,  (size_t)nDis  * D_FEAT);
  fill0_kernel<<<2048, 256, 0, stream>>>(agg_drug, (size_t)nDrug * D_FEAT);

  scatter_add_kernel<<<cdiv_i(E * 32, 256), 256, 0, stream>>>(z_drug, e_src, e_dst, agg_dis, E);
  sage_gemm_kernel<<<cdiv_i(nDis, 64), 128, 0, stream>>>(
      agg_dis, cnt_dis, z_dis, WT[4], WT[5], b2td, z_dis2, nDis, 0);

  scatter_add_kernel<<<cdiv_i(E * 32, 256), 256, 0, stream>>>(z_dis, e_dst, e_src, agg_drug, E);
  sage_gemm_kernel<<<cdiv_i(nDrug, 64), 128, 0, stream>>>(
      agg_drug, cnt_drug, z_drug, WT[6], WT[7], b2dt, z_drug2, nDrug, 0);

  // ---- fused edge classifier ----
  edge_cls_kernel<<<cdiv_i(EL, 64), 128, 0, stream>>>(
      z_drug2, z_dis2, l_src, l_dst, WT[8], bc1, Wc2, bc2, (float*)d_out, EL);
}
